// NATBlock2D_11519102288438
// MI455X (gfx1250) — matrix operand order repaired and checked by reading, not compiled, not run on hardware
//
#include <hip/hip_runtime.h>
#include <math.h>

typedef _Float16 f16;
typedef __attribute__((ext_vector_type(16))) _Float16 v16h;
typedef __attribute__((ext_vector_type(8)))  _Float16 v8h;
typedef __attribute__((ext_vector_type(8)))  float    v8f;
typedef __attribute__((ext_vector_type(4)))  unsigned int u32x4;
typedef __attribute__((ext_vector_type(8)))  int i32x8;
typedef __attribute__((ext_vector_type(4)))  int i32x4;

#define NB   2
#define NHGT 64
#define NWID 64
#define NC   256
#define NHEAD 8
#define HDIM 32
#define KWIN 7
#define NTOK (NB*NHGT*NWID)   // 8192

#if defined(__has_builtin)
# if __has_builtin(__builtin_amdgcn_tensor_load_to_lds)
#  define HAVE_TDM 1
# endif
#endif
#ifndef HAVE_TDM
# define HAVE_TDM 0
#endif

// ---------------- wave32 reductions ----------------
__device__ __forceinline__ float wave_sum(float v) {
#pragma unroll
  for (int o = 16; o > 0; o >>= 1) v += __shfl_xor(v, o, 32);
  return v;
}
__device__ __forceinline__ float wave_max(float v) {
#pragma unroll
  for (int o = 16; o > 0; o >>= 1) v = fmaxf(v, __shfl_xor(v, o, 32));
  return v;
}

#if HAVE_TDM
// ---- Tensor Data Mover: DMA a 16-row x K-halfs weight slab (contiguous) to LDS ----
// D# per CDNA5 ISA ch.8: group0 = {count/type/addr}, group1 = {dims/strides}.
// data_size=1 (2B), 2D tile: tile_dim0=K, tile_dim1=16, tensor_dim0_stride=K.
__device__ __forceinline__ void tdm_load_weights(const f16* gsrc, unsigned lds_off, int K) {
  unsigned long long ga = (unsigned long long)(uintptr_t)gsrc;
  u32x4 g0;
  g0.x = 0x1u;                                              // count=1 (valid user D#)
  g0.y = lds_off;                                           // lds_addr (bytes)
  g0.z = (unsigned)(ga & 0xFFFFFFFFu);                      // global_addr[31:0]
  g0.w = (unsigned)((ga >> 32) & 0x1FFFFFFu) | (2u << 30);  // ga[56:32] | type=2
  i32x8 g1;
  g1[0] = 0x10000;            // workgroup_mask=0, data_size=1 (2 bytes)
  g1[1] = (int)(K << 16);     // tensor_dim0[15:0] = K  (bits 79:48 straddle words)
  g1[2] = (int)(16 << 16);    // tensor_dim0 hi=0, tensor_dim1[15:0] = 16
  g1[3] = (int)(K << 16);     // tensor_dim1 hi=0, tile_dim0 = K
  g1[4] = 16;                 // tile_dim1 = 16, tile_dim2 = 0
  g1[5] = K;                  // tensor_dim0_stride[31:0] = K
  g1[6] = 0;                  // stride hi / tensor_dim1_stride lo
  g1[7] = 0;
  i32x4 gz = {0, 0, 0, 0};
#if __clang_major__ >= 23
  i32x8 gz8 = {0, 0, 0, 0, 0, 0, 0, 0};
  __builtin_amdgcn_tensor_load_to_lds(g0, g1, gz, gz, gz8, 0);
#else
  __builtin_amdgcn_tensor_load_to_lds(g0, g1, gz, gz, 0);
#endif
}
__device__ __forceinline__ void tdm_wait() {
#if __has_builtin(__builtin_amdgcn_s_wait_tensorcnt)
  __builtin_amdgcn_s_wait_tensorcnt(0);
#else
  asm volatile("s_wait_tensorcnt 0x0" ::: "memory");
#endif
}
#endif

// ---------------- f32 -> f16 convert ----------------
__global__ void __launch_bounds__(256) cvt_f32_f16(const float* __restrict__ in,
                                                   f16* __restrict__ out, int n) {
  int i = blockIdx.x * 256 + threadIdx.x;
  if (i < n) out[i] = (f16)in[i];
}

// ---------------- LayerNorm (C=256), wave per token, f16 out ----------------
__global__ void __launch_bounds__(256) ln256_kernel(const float* __restrict__ x,
                                                    const float* __restrict__ g,
                                                    const float* __restrict__ b,
                                                    f16* __restrict__ out) {
  const int wid  = blockIdx.x * 8 + (threadIdx.x >> 5);   // token
  const int lane = threadIdx.x & 31;
  const float* xp = x + (size_t)wid * NC;
  float v[8];
  float s = 0.f;
#pragma unroll
  for (int j = 0; j < 8; ++j) { v[j] = xp[lane + 32 * j]; s += v[j]; }
  const float mean = wave_sum(s) * (1.0f / NC);
  float vs = 0.f;
#pragma unroll
  for (int j = 0; j < 8; ++j) { float d = v[j] - mean; vs += d * d; }
  const float inv = rsqrtf(wave_sum(vs) * (1.0f / NC) + 1e-5f);
  f16* op = out + (size_t)wid * NC;
#pragma unroll
  for (int j = 0; j < 8; ++j) {
    int c = lane + 32 * j;
    op[c] = (f16)((v[j] - mean) * inv * g[c] + b[c]);
  }
}

// ---------------- WMMA GEMM with TDM-staged weights ----------------
// D = A[M,K]f16 * W[N,K]f16^T + bias (+gelu/+resid).
// Block = 8 waves = 128 rows x 16 cols. The 16xK weight slab (contiguous in
// [N,K] layout) is DMA'd to LDS by the Tensor Data Mover once per block and
// shared by all 8 waves through ds_load; A streams direct from global.
template<bool OUT_F16, bool GELU, bool RESID>
__global__ void __launch_bounds__(256) gemm_wmma(const f16* __restrict__ A,
                                                 const f16* __restrict__ Wt,
                                                 const float* __restrict__ bias,
                                                 const float* __restrict__ resid,
                                                 void* __restrict__ outp,
                                                 int M, int N, int K) {
  extern __shared__ f16 ldsW[];                 // 16 * K halfs
  const int lane   = threadIdx.x & 31;
  const int row    = lane & 15;
  const int hi     = lane >> 4;                 // lane-half select
  const int wave   = threadIdx.x >> 5;
  const int tilesN = N >> 4;
  const int n0 = (blockIdx.x % tilesN) << 4;
  const int m0 = ((blockIdx.x / tilesN) << 7) + (wave << 4);

  const f16* wsrc = Wt + (size_t)n0 * K;
#if HAVE_TDM
  if (wave == 0) {
    tdm_load_weights(wsrc, (unsigned)(uintptr_t)&ldsW[0], K);
    tdm_wait();
  }
#else
  for (int idx = threadIdx.x; idx < (16 * K) / 8; idx += 256)
    ((uint4*)ldsW)[idx] = ((const uint4*)wsrc)[idx];
#endif
  __syncthreads();

  // A frag: lane-half hi -> K base hi*16; halfs[0..15]=K+0..15 (contiguous)
  const f16* ap = A + (size_t)(m0 + row) * K + hi * 16;
  // B frag (from LDS): lane-half hi -> halfs[0..15] = K+hi*16 .. +15 (contiguous)
  const f16* bp = ldsW + (size_t)row * K + hi * 16;

  v8f acc = {};
  for (int k0 = 0; k0 < K; k0 += 32) {
    __builtin_prefetch(ap + k0 + 64, 0, 1);
    v8h alo = *(const v8h*)(ap + k0);
    v8h ahi = *(const v8h*)(ap + k0 + 8);
    v16h a = __builtin_shufflevector(alo, ahi,
        0,1,2,3,4,5,6,7,8,9,10,11,12,13,14,15);
    v16h b = *(const v16h*)(bp + k0);
    acc = __builtin_amdgcn_wmma_f32_16x16x32_f16(
        false, a, false, b, (short)0, acc, false, false);
  }

  // D layout: VGPR r, lanes 0-15 -> (M=m0+r, N=n0+lane); lanes 16-31 -> (M=m0+8+r)
  const int n  = n0 + row;
  const float bv = bias[n];
  const int mb = m0 + hi * 8;
#pragma unroll
  for (int r = 0; r < 8; ++r) {
    float v = acc[r] + bv;
    if (GELU) v = 0.5f * v * (1.0f + erff(v * 0.70710678118654752f));
    size_t idx = (size_t)(mb + r) * N + n;
    if (RESID) v += resid[idx];
    if (OUT_F16) ((f16*)outp)[idx] = (f16)v;
    else         ((float*)outp)[idx] = v;
  }
}

// ---------------- Neighborhood attention: wave per (token, head) ----------------
// qkv: [NTOK, 768] f32 (q: 0..255, k: 256..511, v: 512..767; channel = nh*32+d)
// out: [NTOK, 256] f16 (channel = nh*32+d) == transpose(0,2,3,1,4) of reference
__global__ void __launch_bounds__(256) natten_kernel(const float* __restrict__ qkv,
                                                     f16* __restrict__ out) {
  const int wid  = blockIdx.x * 8 + (threadIdx.x >> 5);
  const int lane = threadIdx.x & 31;
  const int nh = wid & (NHEAD - 1);
  const int t  = wid >> 3;            // token
  const int j  = t & 63;
  const int i  = (t >> 6) & 63;
  const int b  = t >> 12;

  const int ch = nh * HDIM + lane;    // this lane's head-dim channel
  const float q = qkv[(size_t)t * 768 + ch] * 0.17677669529663687f; // HD^-0.5

  const int si = min(max(i - 3, 0), NHGT - KWIN);
  const int sj = min(max(j - 3, 0), NWID - KWIN);
  const int tb = b << 12;             // b * 4096

  float l0 = 0.f, l1 = 0.f;
  for (int nb = 0; nb < KWIN * KWIN; ++nb) {
    const int ki = nb / KWIN, kj = nb - ki * KWIN;
    const int tn = tb + (si + ki) * NWID + (sj + kj);
    float d = q * qkv[(size_t)tn * 768 + 256 + ch];
    d = wave_sum(d);
    if (nb == lane)      l0 = d;
    if (nb == lane + 32) l1 = d;
  }

  const bool has1 = (lane < KWIN * KWIN - 32);   // lanes 0..16 hold nb 32..48
  float mx = has1 ? fmaxf(l0, l1) : l0;
  mx = wave_max(mx);
  const float e0 = __expf(l0 - mx);
  const float e1 = has1 ? __expf(l1 - mx) : 0.f;
  const float denom = wave_sum(e0 + e1);
  const float p0 = e0 / denom, p1 = e1 / denom;

  float o = 0.f;
  for (int nb = 0; nb < KWIN * KWIN; ++nb) {
    const int ki = nb / KWIN, kj = nb - ki * KWIN;
    const int tn = tb + (si + ki) * NWID + (sj + kj);
    const float pv = __shfl((nb < 32) ? p0 : p1, nb & 31, 32);
    o += pv * qkv[(size_t)tn * 768 + 512 + ch];
  }
  out[(size_t)t * NC + ch] = (f16)o;
}

// ---------------- launch ----------------
extern "C" void kernel_launch(void* const* d_in, const int* in_sizes, int n_in,
                              void* d_out, int out_size, void* d_ws, size_t ws_size,
                              hipStream_t stream) {
  const float* x      = (const float*)d_in[0];
  const float* ln1_w  = (const float*)d_in[1];
  const float* ln1_b  = (const float*)d_in[2];
  const float* qkv_w  = (const float*)d_in[3];
  const float* qkv_b  = (const float*)d_in[4];
  const float* proj_w = (const float*)d_in[5];
  const float* proj_b = (const float*)d_in[6];
  const float* ln2_w  = (const float*)d_in[7];
  const float* ln2_b  = (const float*)d_in[8];
  const float* fc1_w  = (const float*)d_in[9];
  const float* fc1_b  = (const float*)d_in[10];
  const float* fc2_w  = (const float*)d_in[11];
  const float* fc2_b  = (const float*)d_in[12];
  float* outp = (float*)d_out;

  // workspace carve-up (all region sizes are multiples of 256 bytes)
  char* ws = (char*)d_ws;
  f16*   xn    = (f16*)  ws;                  ws += (size_t)NTOK * NC   * 2;  // 4 MB
  f16*   wq16  = (f16*)  ws;                  ws += (size_t)3*NC  * NC  * 2;  // 384 KB
  f16*   wp16  = (f16*)  ws;                  ws += (size_t)NC    * NC  * 2;  // 128 KB
  f16*   w116  = (f16*)  ws;                  ws += (size_t)4*NC  * NC  * 2;  // 512 KB
  f16*   w216  = (f16*)  ws;                  ws += (size_t)NC  * 4*NC  * 2;  // 512 KB
  float* qkvb  = (float*)ws;                  ws += (size_t)NTOK * 3*NC * 4;  // 24 MB
  f16*   attn  = (f16*)  ws;                  ws += (size_t)NTOK * NC   * 2;  // 4 MB
  float* x2    = (float*)ws;                  ws += (size_t)NTOK * NC   * 4;  // 8 MB
  f16*   h2n   = (f16*)  ws;                  ws += (size_t)NTOK * NC   * 2;  // 4 MB
  f16*   h1    = (f16*)  ws;                  ws += (size_t)NTOK * 4*NC * 2;  // 16 MB

  // 1) weights -> f16
  cvt_f32_f16<<<(3*NC*NC + 255) / 256, 256, 0, stream>>>(qkv_w, wq16, 3*NC*NC);
  cvt_f32_f16<<<(NC*NC   + 255) / 256, 256, 0, stream>>>(proj_w, wp16, NC*NC);
  cvt_f32_f16<<<(4*NC*NC + 255) / 256, 256, 0, stream>>>(fc1_w, w116, 4*NC*NC);
  cvt_f32_f16<<<(4*NC*NC + 255) / 256, 256, 0, stream>>>(fc2_w, w216, 4*NC*NC);

  // 2) LN1
  ln256_kernel<<<NTOK / 8, 256, 0, stream>>>(x, ln1_w, ln1_b, xn);

  // 3) QKV gemm: [8192,256] x [768,256]^T -> f32   (LDS tile = 16*K*2 bytes)
  gemm_wmma<false, false, false>
      <<<(NTOK/128) * (3*NC/16), 256, 16*NC*2, stream>>>(xn, wq16, qkv_b, nullptr,
                                                         qkvb, NTOK, 3*NC, NC);

  // 4) neighborhood attention
  natten_kernel<<<NTOK * NHEAD / 8, 256, 0, stream>>>(qkvb, attn);

  // 5) proj gemm + bias + residual(x) -> x2 (f32)
  gemm_wmma<false, false, true>
      <<<(NTOK/128) * (NC/16), 256, 16*NC*2, stream>>>(attn, wp16, proj_b, x,
                                                       x2, NTOK, NC, NC);

  // 6) LN2
  ln256_kernel<<<NTOK / 8, 256, 0, stream>>>(x2, ln2_w, ln2_b, h2n);

  // 7) FC1 gemm + bias + exact GELU -> f16
  gemm_wmma<true, true, false>
      <<<(NTOK/128) * (4*NC/16), 256, 16*NC*2, stream>>>(h2n, w116, fc1_b, nullptr,
                                                         h1, NTOK, 4*NC, NC);

  // 8) FC2 gemm + bias + residual(x2) -> d_out (f32)   (K=1024 -> 32 KB LDS tile)
  gemm_wmma<false, false, true>
      <<<(NTOK/128) * (NC/16), 256, 16*4*NC*2, stream>>>(h1, w216, fc2_b, x2,
                                                         outp, NTOK, NC, 4*NC);
  (void)in_sizes; (void)n_in; (void)out_size; (void)ws_size;
}
